// Attention_8512625181310
// MI455X (gfx1250) — matrix operand order repaired and checked by reading, not compiled, not run on hardware
//
#include <hip/hip_runtime.h>
#include <hip/hip_bf16.h>

typedef __attribute__((ext_vector_type(16))) _Float16     v16h;
typedef __attribute__((ext_vector_type(8)))  _Float16     v8h;
typedef __attribute__((ext_vector_type(8)))  float        v8f;
typedef __attribute__((ext_vector_type(4)))  unsigned int u32x4;
typedef __attribute__((ext_vector_type(8)))  int          i32x8;
typedef __attribute__((ext_vector_type(4)))  int          i32x4;

#define B_ 4
#define T_ 2048
#define C_ 1024
#define H_ 16
#define D_ 64

#if __has_builtin(__builtin_amdgcn_tensor_load_to_lds)
#define HAVE_TDM 1
#else
#define HAVE_TDM 0
#endif

static __device__ __forceinline__ v8f wmma_f16(v16h a, v16h b, v8f c) {
  return __builtin_amdgcn_wmma_f32_16x16x32_f16(false, a, false, b, (short)0, c,
                                                false, false);
}

// A-fragment (16x32 f16, M x K): lane holds one M-row; lane-half selects K slice.
static __device__ __forceinline__ v16h afrag_h(const _Float16* rowptr, int sel) {
  v16h r;
#pragma unroll
  for (int j = 0; j < 8; ++j) r[j] = rowptr[sel * 8 + j];
#pragma unroll
  for (int j = 0; j < 8; ++j) r[8 + j] = rowptr[16 + sel * 8 + j];
  return r;
}

// B-fragment (32x16 f16, K x N): lane holds one N-column; lane-half selects
// K slice (same order as A-fragment).
static __device__ __forceinline__ v16h bfrag_h(const _Float16* colptr, int kh) {
  v16h r;
#pragma unroll
  for (int j = 0; j < 8; ++j) r[j] = colptr[kh * 8 + j];
#pragma unroll
  for (int j = 0; j < 8; ++j) r[8 + j] = colptr[16 + kh * 8 + j];
  return r;
}

static __device__ __forceinline__ float redmax16(float v) {
  v = fmaxf(v, __shfl_xor(v, 1, 16));
  v = fmaxf(v, __shfl_xor(v, 2, 16));
  v = fmaxf(v, __shfl_xor(v, 4, 16));
  v = fmaxf(v, __shfl_xor(v, 8, 16));
  return v;
}
static __device__ __forceinline__ float redsum16(float v) {
  v += __shfl_xor(v, 1, 16);
  v += __shfl_xor(v, 2, 16);
  v += __shfl_xor(v, 4, 16);
  v += __shfl_xor(v, 8, 16);
  return v;
}

#if HAVE_TDM
// Issue one TDM 2-D tile load (f16 elements) global->LDS per ISA D# layout.
static __device__ __forceinline__ void tdm_load_2d(
    unsigned int lds_byte_addr, const void* gaddr, unsigned int tensor_d0,
    unsigned int tensor_d1, unsigned int tile_d0, unsigned int tile_d1,
    unsigned long long stride0_elems) {
  unsigned long long ga = (unsigned long long)gaddr;
  u32x4 g0;
  g0[0] = 1u;                                              // count=1 (user D#)
  g0[1] = lds_byte_addr;                                   // lds_addr
  g0[2] = (unsigned int)(ga & 0xFFFFFFFFu);                // global_addr[31:0]
  g0[3] = (unsigned int)((ga >> 32) & 0x01FFFFFFu) | 0x80000000u; // [56:32]|type=2
  i32x8 g1;
  g1[0] = (int)(1u << 16);                                 // data_size=1 (2B)
  g1[1] = (int)((tensor_d0 & 0xFFFFu) << 16);              // tensor_dim0[15:0]
  g1[2] = (int)((tensor_d0 >> 16) | ((tensor_d1 & 0xFFFFu) << 16));
  g1[3] = (int)((tensor_d1 >> 16) | (tile_d0 << 16));      // tile_dim0
  g1[4] = (int)(tile_d1 & 0xFFFFu);                        // tile_dim1, tile_dim2=0
  g1[5] = (int)(stride0_elems & 0xFFFFFFFFull);            // dim0_stride[31:0]
  g1[6] = (int)((stride0_elems >> 32) & 0xFFFFull);        // dim0_stride[47:32]
  g1[7] = 0;
  i32x4 z = {0, 0, 0, 0};
#if __clang_major__ >= 23
  i32x8 z8 = {0, 0, 0, 0, 0, 0, 0, 0};
  __builtin_amdgcn_tensor_load_to_lds(g0, g1, z, z, z8, 0);
#else
  __builtin_amdgcn_tensor_load_to_lds(g0, g1, z, z, 0);
#endif
}
#endif

static __device__ __forceinline__ void wait_tensor0() {
#if __has_builtin(__builtin_amdgcn_s_wait_tensorcnt)
  __builtin_amdgcn_s_wait_tensorcnt(0);
#else
  asm volatile("s_wait_tensorcnt 0" ::: "memory");
#endif
}

// ---------------------------------------------------------------------------
// Kernel 0: one-shot f32 -> f16 conversion (8 elements / thread).
// ---------------------------------------------------------------------------
__global__ __launch_bounds__(256) void cvt_f16_kernel(
    const float* __restrict__ src, _Float16* __restrict__ dst, int n8) {
  int i = blockIdx.x * 256 + threadIdx.x;
  if (i >= n8) return;
  int base = i * 8;
  v8h h;
#pragma unroll
  for (int j = 0; j < 8; ++j) h[j] = (_Float16)src[base + j];
  *(v8h*)(dst + base) = h;
}

// ---------------------------------------------------------------------------
// Kernel 1: QKV projection, f16 in / f16 out, f32 accumulate.
// Each wave: 16x64 output strip (A-fragment reused across 4 WMMAs).
//   Q,K stored [b,h,t,d]; V stored transposed [b,h,d,t].
// ---------------------------------------------------------------------------
__global__ __launch_bounds__(128) void qkv_proj_kernel(
    const _Float16* __restrict__ Xh, const _Float16* __restrict__ Whq,
    const _Float16* __restrict__ Whk, const _Float16* __restrict__ Whv,
    const float* __restrict__ bq, const float* __restrict__ bk,
    const float* __restrict__ bv, _Float16* __restrict__ Qh,
    _Float16* __restrict__ Kh, _Float16* __restrict__ Vt) {
  const int lane = threadIdx.x & 31;
  const int wave = threadIdx.x >> 5;
  const int mat  = blockIdx.y;                 // 0=Q 1=K 2=V (uniform)
  const int tile = blockIdx.x * 4 + wave;      // 0..8191
  const int n0   = (tile & 15) * 64;           // 64-channel strip
  const int m0   = (tile >> 4) * 16;           // row (b*T+t) tile

  const _Float16* W    = (mat == 0) ? Whq : (mat == 1) ? Whk : Whv;
  const float*    bias = (mat == 0) ? bq : (mat == 1) ? bk : bv;

  const int nl  = lane & 15;
  const int sel = lane >> 4;

  const _Float16* xrow = Xh + (size_t)(m0 + nl) * C_;
  const _Float16* wr0  = W + (size_t)(n0 + 0 + nl) * C_;
  const _Float16* wr1  = W + (size_t)(n0 + 16 + nl) * C_;
  const _Float16* wr2  = W + (size_t)(n0 + 32 + nl) * C_;
  const _Float16* wr3  = W + (size_t)(n0 + 48 + nl) * C_;

  v8f acc0 = {}, acc1 = {}, acc2 = {}, acc3 = {};
#pragma unroll 4
  for (int k0 = 0; k0 < C_; k0 += 32) {
    v16h a = afrag_h(xrow + k0, sel);
    acc0 = wmma_f16(a, bfrag_h(wr0 + k0, sel), acc0);
    acc1 = wmma_f16(a, bfrag_h(wr1 + k0, sel), acc1);
    acc2 = wmma_f16(a, bfrag_h(wr2 + k0, sel), acc2);
    acc3 = wmma_f16(a, bfrag_h(wr3 + k0, sel), acc3);
  }

#pragma unroll
  for (int c = 0; c < 4; ++c) {
    v8f acc = (c == 0) ? acc0 : (c == 1) ? acc1 : (c == 2) ? acc2 : acc3;
    const float bn = bias[n0 + c * 16 + nl];
    const int ch = n0 + c * 16 + nl;
    const int h = ch >> 6, d = ch & 63;
#pragma unroll
    for (int r = 0; r < 8; ++r) {
      float y = acc[r] + bn;
      int mg = m0 + r + 8 * sel;
      int bb = mg >> 11;
      int t  = mg & (T_ - 1);
      if (mat == 0)
        Qh[(((size_t)(bb * H_ + h)) * T_ + t) * D_ + d] = (_Float16)y;
      else if (mat == 1)
        Kh[(((size_t)(bb * H_ + h)) * T_ + t) * D_ + d] = (_Float16)y;
      else
        Vt[(((size_t)(bb * H_ + h)) * D_ + d) * T_ + t] = (_Float16)y;
    }
  }
}

// ---------------------------------------------------------------------------
// Kernel 2: flash attention. 4 waves/block share one (b,h); K/V tiles are
// staged into LDS by the Tensor Data Mover (double-buffered), then consumed
// by f16 WMMAs with an online softmax.
// ---------------------------------------------------------------------------
__global__ __launch_bounds__(128) void attn_kernel(
    const _Float16* __restrict__ Qh, const _Float16* __restrict__ Kh,
    const _Float16* __restrict__ Vt, float* __restrict__ out) {
  __shared__ _Float16 kbuf[2][32 * 64];  // 32 keys x head-dim 64 (4 KB each)
  __shared__ _Float16 vbuf[2][64 * 32];  // head-dim 64 x 32 keys (4 KB each)
  __shared__ _Float16 pbuf[4][16 * 32];  // per-wave P repack tiles

  const int lane = threadIdx.x & 31;
  const int wave = threadIdx.x >> 5;
  const int nl   = lane & 15;
  const int sel  = lane >> 4;

  const int bh = blockIdx.y;                 // 0..63 (shared by all 4 waves)
  const int qt = blockIdx.x * 4 + wave;      // 0..127
  const int t0 = qt * 16;

  const _Float16* qbase = Qh + (size_t)bh * T_ * D_;
  const _Float16* kbase = Kh + (size_t)bh * T_ * D_;
  const _Float16* vbase = Vt + (size_t)bh * D_ * T_;

  const _Float16* qrow = qbase + (size_t)(t0 + nl) * D_;
  v16h aq0 = afrag_h(qrow, sel);
  v16h aq1 = afrag_h(qrow + 32, sel);

  v8f o0 = {}, o1 = {}, o2 = {}, o3 = {};
  float rmax[8], rsum[8];
#pragma unroll
  for (int r = 0; r < 8; ++r) { rmax[r] = -3.0e38f; rsum[r] = 0.0f; }

  _Float16* lds = &pbuf[wave][0];
  const float scale = 0.125f;  // 1/sqrt(64)

#if HAVE_TDM
  const unsigned int kaddr[2] = {
      (unsigned int)(unsigned long long)&kbuf[0][0],
      (unsigned int)(unsigned long long)&kbuf[1][0]};
  const unsigned int vaddr[2] = {
      (unsigned int)(unsigned long long)&vbuf[0][0],
      (unsigned int)(unsigned long long)&vbuf[1][0]};
  if (wave == 0) {  // prologue: fill buffer 0 for s0 = 0
    tdm_load_2d(kaddr[0], kbase, T_ * D_, 1, 2048, 1, (unsigned long long)T_ * D_);
    tdm_load_2d(vaddr[0], vbase, T_, D_, 32, 64, (unsigned long long)T_);
  }
#endif

  for (int s0 = 0; s0 < T_; s0 += 32) {
    const int cur = (s0 >> 5) & 1;
#if HAVE_TDM
    if (wave == 0) wait_tensor0();   // current K/V tiles resident in LDS
    __syncthreads();                 // publish to all waves; prior reads done
    if (wave == 0 && s0 + 32 < T_) { // overlap next DMA with compute
      const int nxt = cur ^ 1;
      tdm_load_2d(kaddr[nxt], kbase + (size_t)(s0 + 32) * D_, T_ * D_, 1, 2048,
                  1, (unsigned long long)T_ * D_);
      tdm_load_2d(vaddr[nxt], vbase + (s0 + 32), T_, D_, 32, 64,
                  (unsigned long long)T_);
    }
    const _Float16* krow0 = &kbuf[cur][(size_t)nl * D_];
    const _Float16* krow1 = &kbuf[cur][(size_t)(16 + nl) * D_];
    const _Float16* vcol  = &vbuf[cur][(size_t)nl * 32];
    const int vstep = 16 * 32;  // next 16 head-dim columns in LDS tile
#else
    const _Float16* krow0 = kbase + (size_t)(s0 + nl) * D_;
    const _Float16* krow1 = krow0 + 16 * D_;
    const _Float16* vcol  = vbase + (size_t)nl * T_ + s0;
    const int vstep = 16 * T_;
    if (s0 + 32 < T_) __builtin_prefetch(krow0 + 32 * D_, 0, 0);
#endif

    // ---- S = Q K^T for 32 keys (two 16x16 tiles, 4 WMMAs) ----
    v8f sA = {}, sB = {};
    sA = wmma_f16(aq0, bfrag_h(krow0, sel), sA);
    sA = wmma_f16(aq1, bfrag_h(krow0 + 32, sel), sA);
    sB = wmma_f16(aq0, bfrag_h(krow1, sel), sB);
    sB = wmma_f16(aq1, bfrag_h(krow1 + 32, sel), sB);

    // ---- online softmax per row ----
#pragma unroll
    for (int r = 0; r < 8; ++r) {
      float a    = sA[r] * scale;
      float b    = sB[r] * scale;
      float tm   = redmax16(fmaxf(a, b));
      float nm   = fmaxf(rmax[r], tm);
      float corr = __expf(rmax[r] - nm);
      rmax[r]    = nm;
      float p0   = __expf(a - nm);
      float p1   = __expf(b - nm);
      rsum[r]    = rsum[r] * corr + redsum16(p0 + p1);
      o0[r] *= corr; o1[r] *= corr; o2[r] *= corr; o3[r] *= corr;
      int row = r + 8 * sel;
      lds[row * 32 + nl]      = (_Float16)p0;
      lds[row * 32 + 16 + nl] = (_Float16)p1;
    }
    asm volatile("s_wait_dscnt 0" ::: "memory");
    v16h ap = afrag_h(lds + nl * 32, sel);  // P repacked into A layout

    // ---- O += P * V ----
    o0 = wmma_f16(ap, bfrag_h(vcol, sel), o0);
    o1 = wmma_f16(ap, bfrag_h(vcol + 1 * vstep, sel), o1);
    o2 = wmma_f16(ap, bfrag_h(vcol + 2 * vstep, sel), o2);
    o3 = wmma_f16(ap, bfrag_h(vcol + 3 * vstep, sel), o3);
  }

  const int bb = bh >> 4, h = bh & 15;
#pragma unroll
  for (int r = 0; r < 8; ++r) {
    float inv = 1.0f / rsum[r];
    int t = t0 + r + 8 * sel;
    size_t base = ((size_t)(bb * T_ + t)) * C_ + h * D_;
    out[base + 0 + nl]  = o0[r] * inv;
    out[base + 16 + nl] = o1[r] * inv;
    out[base + 32 + nl] = o2[r] * inv;
    out[base + 48 + nl] = o3[r] * inv;
  }
}

extern "C" void kernel_launch(void* const* d_in, const int* in_sizes, int n_in,
                              void* d_out, int out_size, void* d_ws,
                              size_t ws_size, hipStream_t stream) {
  const float* x  = (const float*)d_in[0];
  const float* Wq = (const float*)d_in[1];
  const float* bq = (const float*)d_in[2];
  const float* Wk = (const float*)d_in[3];
  const float* bk = (const float*)d_in[4];
  const float* Wv = (const float*)d_in[5];
  const float* bv = (const float*)d_in[6];
  float* out = (float*)d_out;

  const size_t BHTD = (size_t)B_ * H_ * T_ * D_;  // 8M elements
  const size_t CC   = (size_t)C_ * C_;            // 1M elements
  _Float16* Qh  = (_Float16*)d_ws;
  _Float16* Kh  = Qh + BHTD;
  _Float16* Vt  = Kh + BHTD;
  _Float16* Xh  = Vt + BHTD;
  _Float16* Whq = Xh + (size_t)B_ * T_ * C_;
  _Float16* Whk = Whq + CC;
  _Float16* Whv = Whk + CC;

  // 0) one-shot f32->f16 staging of activations and weights
  cvt_f16_kernel<<<(int)(B_ * T_ * C_ / 8 / 256), 256, 0, stream>>>(x, Xh,
      (int)(B_ * T_ * C_ / 8));
  cvt_f16_kernel<<<(int)(CC / 8 / 256), 256, 0, stream>>>(Wq, Whq, (int)(CC / 8));
  cvt_f16_kernel<<<(int)(CC / 8 / 256), 256, 0, stream>>>(Wk, Whk, (int)(CC / 8));
  cvt_f16_kernel<<<(int)(CC / 8 / 256), 256, 0, stream>>>(Wv, Whv, (int)(CC / 8));

  // 1) QKV projection: 512 M-tiles x 16 N-strips per matrix, 4 waves/block
  dim3 gp((512 * 16) / 4, 3);
  qkv_proj_kernel<<<gp, 128, 0, stream>>>(Xh, Whq, Whk, Whv, bq, bk, bv, Qh, Kh,
                                          Vt);

  // 2) attention: 128 query tiles per (b,h), 4 waves/block
  dim3 ga((T_ / 16) / 4, B_ * H_);
  attn_kernel<<<ga, 128, 0, stream>>>(Qh, Kh, Vt, out);
}
